// SimplifiedLinearAttention_74929999446411
// MI455X (gfx1250) — matrix operand order repaired and checked by reading, not compiled, not run on hardware
//
#include <hip/hip_runtime.h>

// ---------------------------------------------------------------------------
// SimplifiedLinearAttention on MI455X (gfx1250, wave32, WMMA)
//
// Pipeline (all in transposed C-major layout so final reshape is free):
//  0. convert qkv_w / proj_w  f32 -> bf16           (tiny)
//  1. tiled transpose-convert x (B,C,N) -> XT (B,N,C) bf16
//  2. QKV GEMM:  QKV(B,768,N) = W1(768x256) @ X, bf16 WMMA, f32 acc,
//                epilogue: +bias, q=relu, k=relu(k+pos_enc)
//     -- 32x64 tile per wave: each B fragment feeds 2 WMMAs, A reused 4x
//  3. per-(b,h): kv(32x32)=K@V^T, ksum(32)  (f32 VALU, streaming, LDS tiles)
//  4. per-(b,h,n): z = 1/(q.ksum+1e-6); OutT[b,n,h*32+d] = z * sum_c kv[c,d]q[c,n]
//  5. proj GEMM: d_out(B,256,N) = W3(256x256) @ Out, bf16 WMMA, f32 out
// ---------------------------------------------------------------------------

typedef __attribute__((ext_vector_type(16))) __bf16       v16bf;
typedef __attribute__((ext_vector_type(8)))  float        v8f;
typedef __attribute__((ext_vector_type(8)))  unsigned int v8u;
typedef __attribute__((ext_vector_type(4)))  unsigned int v4u;

#define NTOK   6400      // H*W
#define CCH    256       // channels
#define OC_QKV 768
#define NHEAD  8
#define DHEAD  32
#define NBATCH 16

// ---- bf16 <-> f32 helpers (round-to-nearest-even), ushort storage ---------
__device__ __forceinline__ unsigned short f2bfu(float f) {
  unsigned u = __builtin_bit_cast(unsigned, f);
  unsigned r = (u + 0x7FFFu + ((u >> 16) & 1u)) >> 16;
  return (unsigned short)r;
}
__device__ __forceinline__ float bfu2f(unsigned short s) {
  unsigned u = ((unsigned)s) << 16;
  return __builtin_bit_cast(float, u);
}

// Build a v16bf WMMA fragment from two 16-byte chunks (global_load_b128 x2)
__device__ __forceinline__ v16bf ld_frag(const unsigned short* plo,
                                         const unsigned short* phi) {
  v4u lo = *reinterpret_cast<const v4u*>(plo);
  v4u hi = *reinterpret_cast<const v4u*>(phi);
  v8u w  = {lo[0], lo[1], lo[2], lo[3], hi[0], hi[1], hi[2], hi[3]};
  return __builtin_bit_cast(v16bf, w);
}

// ---------------------------------------------------------------------------
// 0) elementwise f32 -> bf16 (weights)
// ---------------------------------------------------------------------------
__global__ void convert_bf16_kernel(const float* __restrict__ src,
                                    unsigned short* __restrict__ dst, int n) {
  int i = blockIdx.x * 256 + threadIdx.x;
  if (i < n) dst[i] = f2bfu(src[i]);
}

// ---------------------------------------------------------------------------
// 1) x (B, C, N) f32 -> XT (B, N, C) bf16, 32x32 tiles through LDS
// ---------------------------------------------------------------------------
__global__ void transpose_convert_kernel(const float* __restrict__ x,
                                         unsigned short* __restrict__ xt) {
  __shared__ float tile[32][33];
  int b  = blockIdx.z;
  int c0 = blockIdx.y * 32;
  int n0 = blockIdx.x * 32;
  int tx = threadIdx.x;   // 32
  int ty = threadIdx.y;   // 8
  const float* xp = x + ((size_t)b * CCH + c0) * NTOK + n0;
#pragma unroll
  for (int i = 0; i < 4; ++i)
    tile[ty + 8 * i][tx] = xp[(size_t)(ty + 8 * i) * NTOK + tx];
  __syncthreads();
  unsigned short* op = xt + ((size_t)b * NTOK + n0) * CCH + c0;
#pragma unroll
  for (int i = 0; i < 4; ++i)
    op[(size_t)(ty + 8 * i) * CCH + tx] = f2bfu(tile[tx][ty + 8 * i]);
}

// ---------------------------------------------------------------------------
// WMMA GEMM core: one wave computes a 32(OC) x 64(N) tile, K = 256.
// A = W (OC x K) bf16 row-major; B = XT (N x K) bf16 row-major.
//   A frag (16x32 bf16): lane<16 -> M=lane, K {k0..k0+7, k0+16..k0+23}
//                        lane>=16 -> M=lane-16, K {k0+8..15, k0+24..31}
//   B frag (32x16 bf16): lane<16 -> n=lane,   K {k0..k0+7, k0+16..k0+23}
//                        lane>=16 -> n=lane-16, K {k0+8..15, k0+24..31}
// Each B fragment is consumed by 2 WMMAs (two M-subtiles): 12 b128 loads
// per K-step drive 8 WMMAs.
// ---------------------------------------------------------------------------
__device__ __forceinline__ void gemm_core32(const unsigned short* __restrict__ W,
                                            const unsigned short* __restrict__ XTb,
                                            int ocBase, int nBase, v8f acc[2][4]) {
  const int lane = threadIdx.x & 31;
  const int half = lane >> 4;
  const int l16  = lane & 15;
  const unsigned short* wrow0 = W + (size_t)(ocBase + l16) * CCH;
  const unsigned short* wrow1 = W + (size_t)(ocBase + 16 + l16) * CCH;
  const unsigned short* xrow  = XTb + (size_t)(nBase + l16) * CCH;

#pragma unroll
  for (int k0 = 0; k0 < CCH; k0 += 32) {
    v16bf a0 = ld_frag(wrow0 + k0 + 8 * half, wrow0 + k0 + 16 + 8 * half);
    v16bf a1 = ld_frag(wrow1 + k0 + 8 * half, wrow1 + k0 + 16 + 8 * half);
#pragma unroll
    for (int t = 0; t < 4; ++t) {
      const unsigned short* xr = xrow + (size_t)t * 16 * CCH + k0;
      v16bf bf = ld_frag(xr + 8 * half, xr + 16 + 8 * half);
      acc[0][t] = __builtin_amdgcn_wmma_f32_16x16x32_bf16(
          false, a0, false, bf, (short)0, acc[0][t], false, false);
      acc[1][t] = __builtin_amdgcn_wmma_f32_16x16x32_bf16(
          false, a1, false, bf, (short)0, acc[1][t], false, false);
    }
  }
}

// ---------------------------------------------------------------------------
// 2) QKV GEMM + bias + relu(q) + relu(k + pos_enc)
//    grid (N/64, 768/256, B), block 256 (8 waves x 32 OC rows = 256 OC rows)
// ---------------------------------------------------------------------------
__global__ __launch_bounds__(256)
void gemm_qkv_kernel(const unsigned short* __restrict__ W1,
                     const unsigned short* __restrict__ XT,
                     const float* __restrict__ bias,
                     const float* __restrict__ pos,
                     unsigned short* __restrict__ QKV) {
  const int b      = blockIdx.z;
  const int wave   = threadIdx.x >> 5;
  const int lane   = threadIdx.x & 31;
  const int half   = lane >> 4;
  const int l16    = lane & 15;
  const int ocBase = blockIdx.y * 256 + wave * 32;
  const int nBase  = blockIdx.x * 64;

  v8f acc[2][4] = {};
  gemm_core32(W1, XT + (size_t)b * NTOK * CCH, ocBase, nBase, acc);

#pragma unroll
  for (int mh = 0; mh < 2; ++mh) {
#pragma unroll
    for (int t = 0; t < 4; ++t) {
      int ncol = nBase + t * 16 + l16;
#pragma unroll
      for (int r = 0; r < 8; ++r) {
        int oc  = ocBase + mh * 16 + 8 * half + r;  // D: lanes>=16 hold M=8..15
        float v = acc[mh][t][r] + bias[oc];
        if (oc < CCH) {                             // q
          v = fmaxf(v, 0.0f);
        } else if (oc < 2 * CCH) {                  // k (+pos_enc, then relu)
          v = fmaxf(v + pos[(size_t)ncol * CCH + (oc - CCH)], 0.0f);
        }                                           // v: passthrough
        QKV[((size_t)b * OC_QKV + oc) * NTOK + ncol] = f2bfu(v);
      }
    }
  }
}

// ---------------------------------------------------------------------------
// 3) per (b,h): kv[c][d] = sum_n k[c,n]*v[d,n];  ksum[c] = sum_n k[c,n]
//    one block per (b,h); LDS tiles n-major to keep global loads coalesced
// ---------------------------------------------------------------------------
__global__ __launch_bounds__(256)
void kv_reduce_kernel(const unsigned short* __restrict__ QKV,
                      float* __restrict__ KV, float* __restrict__ KS) {
  __shared__ float sk[64][33];
  __shared__ float sv[64][33];
  const int bh = blockIdx.x;              // 0..127
  const int b  = bh >> 3, h = bh & 7;
  const unsigned short* Kp = QKV + ((size_t)b * OC_QKV + CCH     + h * DHEAD) * NTOK;
  const unsigned short* Vp = QKV + ((size_t)b * OC_QKV + 2 * CCH + h * DHEAD) * NTOK;

  const int tid  = threadIdx.x;
  const int c    = tid & 31;              // compute mapping
  const int dg   = tid >> 5;              // 8 groups of 4 d's
  const int lrow = tid >> 3;              // load mapping: channel row 0..31
  const int lcol = (tid & 7) * 8;         // 8 consecutive n's

  float a0 = 0.f, a1 = 0.f, a2 = 0.f, a3 = 0.f, ks = 0.f;

  for (int n0 = 0; n0 < NTOK; n0 += 64) {
    v4u kq = *(const v4u*)(Kp + (size_t)lrow * NTOK + n0 + lcol);
    v4u vq = *(const v4u*)(Vp + (size_t)lrow * NTOK + n0 + lcol);
    __syncthreads();
#pragma unroll
    for (int i = 0; i < 8; ++i) {
      unsigned short ku = (unsigned short)((kq[i >> 1] >> ((i & 1) * 16)) & 0xFFFFu);
      unsigned short vu = (unsigned short)((vq[i >> 1] >> ((i & 1) * 16)) & 0xFFFFu);
      sk[lcol + i][lrow] = bfu2f(ku);
      sv[lcol + i][lrow] = bfu2f(vu);
    }
    __syncthreads();
#pragma unroll 4
    for (int n = 0; n < 64; ++n) {
      float kc = sk[n][c];
      if (dg == 0) ks += kc;
      a0 += kc * sv[n][dg * 4 + 0];
      a1 += kc * sv[n][dg * 4 + 1];
      a2 += kc * sv[n][dg * 4 + 2];
      a3 += kc * sv[n][dg * 4 + 3];
    }
  }
  float* kvp = KV + (size_t)bh * 1024 + c * 32 + dg * 4;
  kvp[0] = a0; kvp[1] = a1; kvp[2] = a2; kvp[3] = a3;
  if (dg == 0) KS[bh * 32 + c] = ks;
}

// ---------------------------------------------------------------------------
// 4) z = 1/(q.ksum + 1e-6); OutT[b,n,h*32+d] = z * sum_c kv[c,d]*q[c,n]
//    grid (N/256, B*H); one thread per token column
// ---------------------------------------------------------------------------
__global__ __launch_bounds__(256)
void attn_out_kernel(const unsigned short* __restrict__ QKV,
                     const float* __restrict__ KV, const float* __restrict__ KS,
                     unsigned short* __restrict__ OutT) {
  __shared__ float skv[1024];
  __shared__ float sks[32];
  const int bh = blockIdx.y;
  const int b  = bh >> 3, h = bh & 7;
  const int n  = blockIdx.x * 256 + threadIdx.x;

  for (int i = threadIdx.x; i < 1024; i += 256) skv[i] = KV[(size_t)bh * 1024 + i];
  if (threadIdx.x < 32) sks[threadIdx.x] = KS[bh * 32 + threadIdx.x];
  __syncthreads();

  const unsigned short* Qp = QKV + ((size_t)b * OC_QKV + h * DHEAD) * NTOK;
  float q[32];
  float zden = 0.f;
#pragma unroll
  for (int c = 0; c < 32; ++c) {
    q[c] = bfu2f(Qp[(size_t)c * NTOK + n]);
    zden += q[c] * sks[c];
  }
  float z = 1.0f / (zden + 1e-6f);

  unsigned short* op = OutT + ((size_t)b * NTOK + n) * CCH + h * DHEAD;
#pragma unroll
  for (int d = 0; d < 32; ++d) {
    float a = 0.f;
#pragma unroll
    for (int c = 0; c < 32; ++c) a += skv[c * 32 + d] * q[c];
    op[d] = f2bfu(a * z);
  }
}

// ---------------------------------------------------------------------------
// 5) proj GEMM: d_out(B,256,N) = W3 @ Out + proj_b  (f32 output, final layout)
//    grid (N/64, 1, B)
// ---------------------------------------------------------------------------
__global__ __launch_bounds__(256)
void gemm_proj_kernel(const unsigned short* __restrict__ W3,
                      const unsigned short* __restrict__ OutT,
                      const float* __restrict__ bias,
                      float* __restrict__ Y) {
  const int b      = blockIdx.z;
  const int wave   = threadIdx.x >> 5;
  const int lane   = threadIdx.x & 31;
  const int half   = lane >> 4;
  const int l16    = lane & 15;
  const int ocBase = wave * 32;            // 8 waves cover all 256 OC rows
  const int nBase  = blockIdx.x * 64;

  v8f acc[2][4] = {};
  gemm_core32(W3, OutT + (size_t)b * NTOK * CCH, ocBase, nBase, acc);

#pragma unroll
  for (int mh = 0; mh < 2; ++mh) {
#pragma unroll
    for (int t = 0; t < 4; ++t) {
      int ncol = nBase + t * 16 + l16;
#pragma unroll
      for (int r = 0; r < 8; ++r) {
        int oc = ocBase + mh * 16 + 8 * half + r;
        Y[((size_t)b * CCH + oc) * NTOK + ncol] = acc[mh][t][r] + bias[oc];
      }
    }
  }
}

// ---------------------------------------------------------------------------
// launcher
// ---------------------------------------------------------------------------
extern "C" void kernel_launch(void* const* d_in, const int* in_sizes, int n_in,
                              void* d_out, int out_size, void* d_ws, size_t ws_size,
                              hipStream_t stream) {
  (void)in_sizes; (void)n_in; (void)out_size; (void)ws_size;
  const float* x     = (const float*)d_in[0];   // (16,256,80,80)
  const float* qkv_w = (const float*)d_in[1];   // (768,256)
  const float* qkv_b = (const float*)d_in[2];   // (768)
  const float* pos   = (const float*)d_in[3];   // (1,6400,256)
  const float* pw    = (const float*)d_in[4];   // (256,256)
  const float* pb    = (const float*)d_in[5];   // (256)

  char* ws = (char*)d_ws;
  // workspace map (all offsets 256B aligned), total ~251 MB
  unsigned short* XT   = (unsigned short*)(ws);               //  52,428,800 B
  unsigned short* QKV  = (unsigned short*)(ws +  52428800);   // 157,286,400 B
  unsigned short* OutT = (unsigned short*)(ws + 209715200);   //  52,428,800 B
  unsigned short* W1   = (unsigned short*)(ws + 262144000);   //     393,216 B
  unsigned short* W3   = (unsigned short*)(ws + 262537216);   //     131,072 B
  float*          KV   = (float*)        (ws + 262668288);    //     524,288 B
  float*          KS   = (float*)        (ws + 263192576);    //      16,384 B

  convert_bf16_kernel<<<(OC_QKV * CCH + 255) / 256, 256, 0, stream>>>(qkv_w, W1, OC_QKV * CCH);
  convert_bf16_kernel<<<(CCH * CCH + 255) / 256, 256, 0, stream>>>(pw, W3, CCH * CCH);

  transpose_convert_kernel<<<dim3(NTOK / 32, CCH / 32, NBATCH), dim3(32, 8), 0, stream>>>(x, XT);

  gemm_qkv_kernel<<<dim3(NTOK / 64, OC_QKV / 256, NBATCH), 256, 0, stream>>>(W1, XT, qkv_b, pos, QKV);

  kv_reduce_kernel<<<NBATCH * NHEAD, 256, 0, stream>>>(QKV, KV, KS);

  attn_out_kernel<<<dim3(NTOK / 256, NBATCH * NHEAD), 256, 0, stream>>>(QKV, KV, KS, OutT);

  gemm_proj_kernel<<<dim3(NTOK / 64, 1, NBATCH), 256, 0, stream>>>(W3, OutT, pb, (float*)d_out);
}
